// GlobalTransferer_54571854462990
// MI455X (gfx1250) — matrix operand order repaired and checked by reading, not compiled, not run on hardware
//
#include <hip/hip_runtime.h>

// ---------------------------------------------------------------------------
// CDNA5 (gfx1250, wave32) coarse-to-fine NN matching.
// dist = |a|^2 - 2*q.a  (|q|^2 constant per argmin row -> dropped)
// q.a via split-bf16 (hi+lo): hi*hi + hi*lo + lo*hi on v_wmma_f32_16x16x32_bf16
// -> ~fp32 accuracy at 3x bf16 cost (vs 8x for the f32 WMMA path).
//
// Fast path (workspace permitting): a one-touch pre-pass splits every element
// into [row][ hi[K] | lo[K] ] bf16 in d_ws and computes exact f32 row norms,
// so the gram hot loops are pure loads + WMMA (no per-use conversions, which
// otherwise cost ~64x-redundant VALU work). Fine kernel stages its strided
// 4x4x1KB query tile into LDS with the Tensor Data Mover.
// ---------------------------------------------------------------------------

typedef __attribute__((ext_vector_type(16))) __bf16 v16bf;
typedef __attribute__((ext_vector_type(8)))  __bf16 v8bf;
typedef __attribute__((ext_vector_type(8)))  float  v8f;
typedef __attribute__((ext_vector_type(4)))  unsigned int u32x4;
typedef __attribute__((ext_vector_type(8)))  int i32x8;
typedef __attribute__((ext_vector_type(4)))  int i32x4;

#define FLT_BIG 3.4028234663852886e+38f

#ifndef __has_builtin
#define __has_builtin(x) 0
#endif
#if __has_builtin(__builtin_amdgcn_tensor_load_to_lds) && \
    __has_builtin(__builtin_amdgcn_s_wait_tensorcnt)
#define HAVE_TDM 1
#else
#define HAVE_TDM 0
#endif

// ======================= shared helpers =======================

__device__ inline v16bf cat8(v8bf a, v8bf b) {
  return __builtin_shufflevector(a, b, 0, 1, 2, 3, 4, 5, 6, 7,
                                 8, 9, 10, 11, 12, 13, 14, 15);
}

// Pre-split fragment loads. Row layout: hi[stride] then lo[stride] bf16.
// A-frag (16-bit A 16x32): 16 contiguous at kb,
// kb already includes (lane>=16 ? 16 : 0).
__device__ inline void loadFragA_hl(const __bf16* __restrict__ base, int kb,
                                    int stride, v16bf& hi, v16bf& lo) {
  hi = cat8(*(const v8bf*)(base + kb), *(const v8bf*)(base + kb + 8));
  lo = cat8(*(const v8bf*)(base + stride + kb),
            *(const v8bf*)(base + stride + kb + 8));
}

// B-frag (16-bit B 32x16): 16 contiguous at kb (kb includes lane>=16 ? 16 : 0).
__device__ inline void loadFragB_hl(const __bf16* __restrict__ base, int kb,
                                    int stride, v16bf& hi, v16bf& lo) {
  const v8bf* p = (const v8bf*)(base + kb);
  hi = cat8(p[0], p[1]);
  const v8bf* q = (const v8bf*)(base + stride + kb);
  lo = cat8(q[0], q[1]);
}

__device__ inline v8f wmma3v(v16bf ah, v16bf al, v16bf bh, v16bf bl, v8f c) {
  c = __builtin_amdgcn_wmma_f32_16x16x32_bf16(false, ah, false, bh,
                                              (short)0, c, false, false);
  c = __builtin_amdgcn_wmma_f32_16x16x32_bf16(false, ah, false, bl,
                                              (short)0, c, false, false);
  c = __builtin_amdgcn_wmma_f32_16x16x32_bf16(false, al, false, bh,
                                              (short)0, c, false, false);
  return c;
}

// ======================= direct-path (fallback) helpers =======================

struct Frag { v16bf hi; v16bf lo; };

__device__ inline void split16(const float* __restrict__ x, Frag& f) {
#pragma unroll
  for (int i = 0; i < 16; ++i) {
    float v = x[i];
    __bf16 h = (__bf16)v;
    f.hi[i] = h;
    f.lo[i] = (__bf16)(v - (float)h);
  }
}

__device__ inline Frag loadA_f32(const float* __restrict__ row, int kbase) {
  const float4* p = (const float4*)(row + kbase);
  const float4* q = (const float4*)(row + kbase + 8);
  float4 a0 = p[0], a1 = p[1], b0 = q[0], b1 = q[1];
  float x[16] = {a0.x, a0.y, a0.z, a0.w, a1.x, a1.y, a1.z, a1.w,
                 b0.x, b0.y, b0.z, b0.w, b1.x, b1.y, b1.z, b1.w};
  Frag f;
  split16(x, f);
  return f;
}

__device__ inline Frag loadB_f32(const float* __restrict__ row, int kbase,
                                 float& sq) {
  const float4* p = (const float4*)(row + kbase);
  float4 x0 = p[0], x1 = p[1], x2 = p[2], x3 = p[3];
  float x[16] = {x0.x, x0.y, x0.z, x0.w, x1.x, x1.y, x1.z, x1.w,
                 x2.x, x2.y, x2.z, x2.w, x3.x, x3.y, x3.z, x3.w};
#pragma unroll
  for (int i = 0; i < 16; ++i) sq += x[i] * x[i];
  Frag f;
  split16(x, f);
  return f;
}

__device__ inline v8f wmma3(const Frag& a, const Frag& b, v8f c) {
  return wmma3v(a.hi, a.lo, b.hi, b.lo, c);
}

// ======================= pre-split conversion pass =======================
// src: [R][K] f32  ->  dst: [R][2K] bf16 (hi[K] | lo[K]);  nrm[R] = sum(x^2)
__global__ __launch_bounds__(256) void convert_split_kernel(
    const float* __restrict__ src, __bf16* __restrict__ dst,
    float* __restrict__ nrm, int K) {
  const int row = blockIdx.x;
  const int lane = threadIdx.x & 31;
  const int wave = threadIdx.x >> 5;
  const float* s = src + (size_t)row * K;
  __bf16* d = dst + (size_t)row * (2 * K);
  float sq = 0.f;
  for (int k = threadIdx.x; k < K; k += 256) {
    float v = s[k];
    __bf16 h = (__bf16)v;
    d[k] = h;
    d[K + k] = (__bf16)(v - (float)h);
    sq += v * v;
  }
#pragma unroll
  for (int dlt = 16; dlt >= 1; dlt >>= 1) sq += __shfl_xor(sq, dlt, 32);
  __shared__ float red[8];
  if (lane == 0) red[wave] = sq;
  __syncthreads();
  if (threadIdx.x == 0) {
    float t = 0.f;
#pragma unroll
    for (int w = 0; w < 8; ++w) t += red[w];
    nrm[row] = t;
  }
}

// ======================= coarse matching (fast path) =======================
// B=4, 1024 rows, K=1024. grid=(32 rowPairs, 4 batches), block=256 (8 waves).
// Wave tile = 32 rows x 32 cols (2x2 WMMA tiles) -> halves L2 re-read traffic.
__global__ __launch_bounds__(256) void coarse_match_ws(
    const __bf16* __restrict__ Whl, const __bf16* __restrict__ Ahl,
    const float* __restrict__ Anrm, int* __restrict__ outCoarse) {
  const int lane = threadIdx.x & 31;
  const int wave = threadIdx.x >> 5;
  const int l16 = lane & 15;
  const int h16 = lane >> 4;
  const int b = blockIdx.y;
  const int rowPair = blockIdx.x;  // 0..31 -> 32 query rows

  const __bf16* Wb = Whl + (size_t)b * 1024 * 2048;
  const __bf16* Ab = Ahl + (size_t)b * 1024 * 2048;
  const float* Nb = Anrm + b * 1024;

  const __bf16* qr0 = Wb + (size_t)(rowPair * 32 + l16) * 2048;
  const __bf16* qr1 = qr0 + (size_t)16 * 2048;

  float bestVal[16];
  int bestIdx[16];
#pragma unroll
  for (int r = 0; r < 16; ++r) { bestVal[r] = FLT_BIG; bestIdx[r] = 0x7fffffff; }

  for (int cp = wave; cp < 32; cp += 8) {
    const int n0 = cp * 32 + l16, n1 = n0 + 16;
    const __bf16* ar0 = Ab + (size_t)n0 * 2048;
    const __bf16* ar1 = Ab + (size_t)n1 * 2048;
    if (cp + 8 < 32) {  // hint the next column pair this wave will sweep
      __builtin_prefetch(Ab + (size_t)(n0 + 256) * 2048, 0, 1);
      __builtin_prefetch(Ab + (size_t)(n1 + 256) * 2048, 0, 1);
    }
    const float nrm0 = Nb[n0], nrm1 = Nb[n1];
    v8f c00 = {0.f, 0.f, 0.f, 0.f, 0.f, 0.f, 0.f, 0.f};
    v8f c01 = c00, c10 = c00, c11 = c00;
#pragma unroll 2
    for (int kc = 0; kc < 32; ++kc) {
      const int kbA = kc * 32 + h16 * 16;
      const int kbB = kc * 32 + h16 * 16;
      v16bf a0h, a0l, a1h, a1l, b0h, b0l, b1h, b1l;
      loadFragA_hl(qr0, kbA, 1024, a0h, a0l);
      loadFragA_hl(qr1, kbA, 1024, a1h, a1l);
      loadFragB_hl(ar0, kbB, 1024, b0h, b0l);
      loadFragB_hl(ar1, kbB, 1024, b1h, b1l);
      c00 = wmma3v(a0h, a0l, b0h, b0l, c00);
      c01 = wmma3v(a0h, a0l, b1h, b1l, c01);
      c10 = wmma3v(a1h, a1l, b0h, b0l, c10);
      c11 = wmma3v(a1h, a1l, b1h, b1l, c11);
    }
#pragma unroll
    for (int r = 0; r < 8; ++r) {
      float s;
      s = nrm0 - 2.f * c00[r];
      if (s < bestVal[r] || (s == bestVal[r] && n0 < bestIdx[r])) { bestVal[r] = s; bestIdx[r] = n0; }
      s = nrm1 - 2.f * c01[r];
      if (s < bestVal[r] || (s == bestVal[r] && n1 < bestIdx[r])) { bestVal[r] = s; bestIdx[r] = n1; }
      s = nrm0 - 2.f * c10[r];
      if (s < bestVal[8 + r] || (s == bestVal[8 + r] && n0 < bestIdx[8 + r])) { bestVal[8 + r] = s; bestIdx[8 + r] = n0; }
      s = nrm1 - 2.f * c11[r];
      if (s < bestVal[8 + r] || (s == bestVal[8 + r] && n1 < bestIdx[8 + r])) { bestVal[8 + r] = s; bestIdx[8 + r] = n1; }
    }
  }

  __shared__ float sVal[8][16][32];
  __shared__ int sIdx[8][16][32];
#pragma unroll
  for (int r = 0; r < 16; ++r) {
    sVal[wave][r][lane] = bestVal[r];
    sIdx[wave][r][lane] = bestIdx[r];
  }
  __syncthreads();

  if (threadIdx.x < 32) {
    const int t = threadIdx.x;          // local row 0..31
    const int tile = t >> 4;
    const int rowIn = t & 15;
    const int half = rowIn >> 3, r = rowIn & 7;
    const int slot = tile * 8 + r;
    float bv = FLT_BIG;
    int bi = 0x7fffffff;
    for (int w = 0; w < 8; ++w)
      for (int li = 0; li < 16; ++li) {
        const float v = sVal[w][slot][half * 16 + li];
        const int ix = sIdx[w][slot][half * 16 + li];
        if (v < bv || (v == bv && ix < bi)) { bv = v; bi = ix; }
      }
    const int row = rowPair * 32 + t;
    int* o = outCoarse + ((size_t)b * 1024 + row) * 2;
    o[0] = bi >> 5;
    o[1] = bi & 31;
  }
}

// ======================= fine matching (fast path) =======================
// 4096 cells, block=128 (4 waves). 16 queries x 144 candidates, K=256.
// Query tile (strided 4x4 x 1KB) staged into LDS by the Tensor Data Mover.
__global__ __launch_bounds__(128) void fine_match_ws(
    const __bf16* __restrict__ Ahl, const __bf16* __restrict__ Whl,
    const float* __restrict__ Anrm, const int* __restrict__ cm,
    int* __restrict__ outFine) {
  const int lane = threadIdx.x & 31;
  const int wave = threadIdx.x >> 5;
  const int l16 = lane & 15;
  const int h16 = lane >> 4;
  const int cell = blockIdx.x;
  const int b = cell >> 10;
  const int ci = (cell >> 5) & 31;
  const int cj = cell & 31;

  __shared__ __align__(16) __bf16 qtile[16 * 512];  // 16 pos x (hi[256]|lo[256])
  __shared__ int nbr[9][2];

  // global start of the 4x4 query-position tile (pos stride = 512 bf16 = 1KB)
  const __bf16* wb =
      Whl + ((size_t)b * 16384 + (size_t)(ci * 4) * 128 + (size_t)(cj * 4)) * 512;

  if (wave == 0) {
#if HAVE_TDM
    // D#: data_size=4B; tensor dims (dwords): dim0=256 (per-pos payload),
    // dim1=128 (col, stride 256), dim2=128 (row, stride 32768);
    // tile 256 x 4 x 4 -> lands contiguous in LDS in pos-major order.
    const unsigned long long ga = (unsigned long long)(size_t)wb;
    const unsigned int la = (unsigned int)(size_t)&qtile[0];
    u32x4 g0;
    g0[0] = 1u;                                   // count=1
    g0[1] = la;                                   // lds_addr
    g0[2] = (unsigned int)ga;                     // global_addr[31:0]
    g0[3] = (unsigned int)((ga >> 32) & 0x01FFFFFFu) | (2u << 30);  // addr hi | type=2
    i32x8 g1;
    g1[0] = (2 << 16);                            // data_size = 4B
    g1[1] = (int)(256u << 16);                    // tensor_dim0[15:0]
    g1[2] = (int)(128u << 16);                    // tensor_dim1[15:0]
    g1[3] = (int)(256u << 16);                    // tile_dim0
    g1[4] = (4) | (4 << 16);                      // tile_dim1, tile_dim2
    g1[5] = 256;                                  // tensor_dim0_stride[31:0]
    g1[6] = (int)0x80000000u;                     // dim1_stride[15:0]=32768 << 16
    g1[7] = 0;
    i32x4 g2; g2[0] = 128; g2[1] = 0; g2[2] = 0; g2[3] = 0;  // tensor_dim2
    i32x4 g3; g3[0] = 0; g3[1] = 0; g3[2] = 0; g3[3] = 0;
#if __clang_major__ >= 23
    i32x8 g4 = {0, 0, 0, 0, 0, 0, 0, 0};
    __builtin_amdgcn_tensor_load_to_lds(g0, g1, g2, g3, g4, 0);
#else
    __builtin_amdgcn_tensor_load_to_lds(g0, g1, g2, g3, 0);
#endif
    __builtin_amdgcn_s_wait_tensorcnt(0);
#else
    // manual staged copy (16 pos x 64 x 16B)
    for (int e = lane; e < 16 * 64; e += 32) {
      const int p = e >> 6, u = e & 63;
      const float4* s4 =
          (const float4*)(wb + ((size_t)((p >> 2) * 128 + (p & 3))) * 512);
      ((float4*)qtile)[p * 64 + u] = s4[u];
    }
#endif
  }
  if (wave == 1 && lane < 9) {
    const int di = lane / 3 - 1, dj = lane % 3 - 1;
    const int ii = min(max(ci + di, 0), 31);
    const int jj = min(max(cj + dj, 0), 31);
    const int* c2 = cm + (((size_t)b * 32 + ii) * 32 + jj) * 2;
    nbr[lane][0] = c2[0];
    nbr[lane][1] = c2[1];
  }
  __syncthreads();

  const __bf16* Ab = Ahl + (size_t)b * 16384 * 512;
  const float* Nb = Anrm + b * 16384;
  const __bf16* qbase = qtile + l16 * 512;  // query p = l16, in LDS

  float bestVal[8];
  int bestIdx[8];
#pragma unroll
  for (int r = 0; r < 8; ++r) { bestVal[r] = FLT_BIG; bestIdx[r] = 0x7fffffff; }

  for (int ct = wave; ct < 9; ct += 4) {
    const int cr = nbr[ct][0] * 4 + (l16 >> 2);
    const int cc = nbr[ct][1] * 4 + (l16 & 3);
    const __bf16* arow = Ab + ((size_t)cr * 128 + cc) * 512;
    const float nrm = Nb[cr * 128 + cc];
    v8f c = {0.f, 0.f, 0.f, 0.f, 0.f, 0.f, 0.f, 0.f};
#pragma unroll
    for (int kc = 0; kc < 8; ++kc) {
      v16bf ah, al, bh, bl;
      loadFragA_hl(qbase, kc * 32 + h16 * 16, 256, ah, al);   // ds_load from LDS
      loadFragB_hl(arow, kc * 32 + h16 * 16, 256, bh, bl);   // global gather
      c = wmma3v(ah, al, bh, bl, c);
    }
    const int n = ct * 16 + l16;
#pragma unroll
    for (int r = 0; r < 8; ++r) {
      const float s = nrm - 2.0f * c[r];
      if (s < bestVal[r] || (s == bestVal[r] && n < bestIdx[r])) {
        bestVal[r] = s;
        bestIdx[r] = n;
      }
    }
  }

  __shared__ float sVal[4][8][32];
  __shared__ int sIdx[4][8][32];
#pragma unroll
  for (int r = 0; r < 8; ++r) {
    sVal[wave][r][lane] = bestVal[r];
    sIdx[wave][r][lane] = bestIdx[r];
  }
  __syncthreads();

  if (threadIdx.x < 16) {
    const int p = threadIdx.x;
    const int half = p >> 3, r = p & 7;
    float bv = FLT_BIG;
    int bi = 0x7fffffff;
    for (int w = 0; w < 4; ++w)
      for (int li = 0; li < 16; ++li) {
        const float v = sVal[w][r][half * 16 + li];
        const int ix = sIdx[w][r][half * 16 + li];
        if (v < bv || (v == bv && ix < bi)) { bv = v; bi = ix; }
      }
    const int ct = bi >> 4, s = bi & 15;
    const int br = nbr[ct][0] * 4 + (s >> 2);
    const int bc = nbr[ct][1] * 4 + (s & 3);
    const int fr = ci * 4 + (p >> 2);
    const int fc = cj * 4 + (p & 3);
    int* o = outFine + (((size_t)b * 128 + fr) * 128 + fc) * 2;
    o[0] = br;
    o[1] = bc;
  }
}

// ======================= direct (no-workspace) fallback kernels =======================

__global__ __launch_bounds__(256) void coarse_match_direct(
    const float* __restrict__ f1, const float* __restrict__ fk,
    int* __restrict__ outCoarse) {
  const int lane = threadIdx.x & 31;
  const int wave = threadIdx.x >> 5;
  const int l16 = lane & 15;
  const int h16 = lane >> 4;
  const int b = blockIdx.y;
  const int rowTile = blockIdx.x;

  const float* Q = fk + (size_t)b * 1024 * 1024;
  const float* A = f1 + (size_t)b * 1024 * 1024;
  const float* qrow = Q + ((size_t)(rowTile * 16 + l16)) * 1024;

  float bestVal[8];
  int bestIdx[8];
#pragma unroll
  for (int r = 0; r < 8; ++r) { bestVal[r] = FLT_BIG; bestIdx[r] = 0x7fffffff; }

  for (int ct = wave; ct < 64; ct += 8) {
    const int n = ct * 16 + l16;
    const float* arow = A + (size_t)n * 1024;
    v8f c = {0.f, 0.f, 0.f, 0.f, 0.f, 0.f, 0.f, 0.f};
    float sq = 0.f;
#pragma unroll 4
    for (int kc = 0; kc < 32; ++kc) {
      Frag fa = loadA_f32(qrow, kc * 32 + h16 * 16);
      Frag fb = loadB_f32(arow, kc * 32 + h16 * 16, sq);
      c = wmma3(fa, fb, c);
    }
    const float nrm = sq + __shfl_xor(sq, 16, 32);
#pragma unroll
    for (int r = 0; r < 8; ++r) {
      const float s = nrm - 2.0f * c[r];
      if (s < bestVal[r] || (s == bestVal[r] && n < bestIdx[r])) {
        bestVal[r] = s;
        bestIdx[r] = n;
      }
    }
  }

  __shared__ float sVal[8][8][32];
  __shared__ int sIdx[8][8][32];
#pragma unroll
  for (int r = 0; r < 8; ++r) {
    sVal[wave][r][lane] = bestVal[r];
    sIdx[wave][r][lane] = bestIdx[r];
  }
  __syncthreads();

  if (threadIdx.x < 16) {
    const int p = threadIdx.x;
    const int half = p >> 3, r = p & 7;
    float bv = FLT_BIG;
    int bi = 0x7fffffff;
    for (int w = 0; w < 8; ++w)
      for (int li = 0; li < 16; ++li) {
        const float v = sVal[w][r][half * 16 + li];
        const int ix = sIdx[w][r][half * 16 + li];
        if (v < bv || (v == bv && ix < bi)) { bv = v; bi = ix; }
      }
    const int row = rowTile * 16 + p;
    int* o = outCoarse + ((size_t)b * 1024 + row) * 2;
    o[0] = bi >> 5;
    o[1] = bi & 31;
  }
}

__global__ __launch_bounds__(128) void fine_match_direct(
    const float* __restrict__ f1, const float* __restrict__ fk,
    const int* __restrict__ cm, int* __restrict__ outFine) {
  const int lane = threadIdx.x & 31;
  const int wave = threadIdx.x >> 5;
  const int l16 = lane & 15;
  const int h16 = lane >> 4;
  const int cell = blockIdx.x;
  const int b = cell >> 10;
  const int ci = (cell >> 5) & 31;
  const int cj = cell & 31;

  __shared__ int nbr[9][2];
  if (threadIdx.x < 9) {
    const int di = (int)threadIdx.x / 3 - 1;
    const int dj = (int)threadIdx.x % 3 - 1;
    const int ii = min(max(ci + di, 0), 31);
    const int jj = min(max(cj + dj, 0), 31);
    const int* c2 = cm + (((size_t)b * 32 + ii) * 32 + jj) * 2;
    nbr[threadIdx.x][0] = c2[0];
    nbr[threadIdx.x][1] = c2[1];
  }
  __syncthreads();

  const float* F1 = f1 + (size_t)b * 128 * 128 * 256;
  const float* FK = fk + (size_t)b * 128 * 128 * 256;
  const int qr = ci * 4 + (l16 >> 2);
  const int qc = cj * 4 + (l16 & 3);
  const float* qrow = FK + ((size_t)qr * 128 + qc) * 256;

  float bestVal[8];
  int bestIdx[8];
#pragma unroll
  for (int r = 0; r < 8; ++r) { bestVal[r] = FLT_BIG; bestIdx[r] = 0x7fffffff; }

  for (int ct = wave; ct < 9; ct += 4) {
    const int cr = nbr[ct][0] * 4 + (l16 >> 2);
    const int cc = nbr[ct][1] * 4 + (l16 & 3);
    const float* arow = F1 + ((size_t)cr * 128 + cc) * 256;
    v8f c = {0.f, 0.f, 0.f, 0.f, 0.f, 0.f, 0.f, 0.f};
    float sq = 0.f;
#pragma unroll
    for (int kc = 0; kc < 8; ++kc) {
      Frag fa = loadA_f32(qrow, kc * 32 + h16 * 16);
      Frag fb = loadB_f32(arow, kc * 32 + h16 * 16, sq);
      c = wmma3(fa, fb, c);
    }
    const float nrm = sq + __shfl_xor(sq, 16, 32);
    const int n = ct * 16 + l16;
#pragma unroll
    for (int r = 0; r < 8; ++r) {
      const float s = nrm - 2.0f * c[r];
      if (s < bestVal[r] || (s == bestVal[r] && n < bestIdx[r])) {
        bestVal[r] = s;
        bestIdx[r] = n;
      }
    }
  }

  __shared__ float sVal[4][8][32];
  __shared__ int sIdx[4][8][32];
#pragma unroll
  for (int r = 0; r < 8; ++r) {
    sVal[wave][r][lane] = bestVal[r];
    sIdx[wave][r][lane] = bestIdx[r];
  }
  __syncthreads();

  if (threadIdx.x < 16) {
    const int p = threadIdx.x;
    const int half = p >> 3, r = p & 7;
    float bv = FLT_BIG;
    int bi = 0x7fffffff;
    for (int w = 0; w < 4; ++w)
      for (int li = 0; li < 16; ++li) {
        const float v = sVal[w][r][half * 16 + li];
        const int ix = sIdx[w][r][half * 16 + li];
        if (v < bv || (v == bv && ix < bi)) { bv = v; bi = ix; }
      }
    const int ct = bi >> 4, s = bi & 15;
    const int br = nbr[ct][0] * 4 + (s >> 2);
    const int bc = nbr[ct][1] * 4 + (s & 3);
    const int fr = ci * 4 + (p >> 2);
    const int fc = cj * 4 + (p & 3);
    int* o = outFine + (((size_t)b * 128 + fr) * 128 + fc) * 2;
    o[0] = br;
    o[1] = bc;
  }
}

// ======================= host launcher =======================

extern "C" void kernel_launch(void* const* d_in, const int* in_sizes, int n_in,
                              void* d_out, int out_size, void* d_ws, size_t ws_size,
                              hipStream_t stream) {
  (void)in_sizes; (void)n_in; (void)out_size;
  const float* f1c = (const float*)d_in[0];  // F_g1_coarse (4,32,32,1024)  db
  const float* fkc = (const float*)d_in[1];  // F_gk_coarse (4,32,32,1024)  queries
  const float* f1f = (const float*)d_in[2];  // F_g1_fine   (4,128,128,256) db
  const float* fkf = (const float*)d_in[3];  // F_gk_fine   (4,128,128,256) queries

  int* out = (int*)d_out;
  int* outCoarse = out;                    // (4,32,32,2)
  int* outFine = out + 4 * 32 * 32 * 2;    // (4,128,128,2)

  // workspace carve-up for the pre-split fast path
  char* ws = (char*)d_ws;
  size_t off = 0;
  auto carve = [&](size_t bytes) -> void* {
    void* p = ws + off;
    off += (bytes + 255) & ~(size_t)255;
    return p;
  };
  __bf16* Ac = (__bf16*)carve((size_t)4096 * 2048 * 2);   // coarse db   hi|lo
  __bf16* Wc = (__bf16*)carve((size_t)4096 * 2048 * 2);   // coarse qry  hi|lo
  __bf16* Af = (__bf16*)carve((size_t)65536 * 512 * 2);   // fine db     hi|lo
  __bf16* Wf = (__bf16*)carve((size_t)65536 * 512 * 2);   // fine qry    hi|lo
  float* nAc = (float*)carve((size_t)4096 * 4);
  float* nWc = (float*)carve((size_t)4096 * 4);
  float* nAf = (float*)carve((size_t)65536 * 4);
  float* nWf = (float*)carve((size_t)65536 * 4);
  const size_t need = off;

  if (ws_size >= need) {
    convert_split_kernel<<<4096, 256, 0, stream>>>(f1c, Ac, nAc, 1024);
    convert_split_kernel<<<4096, 256, 0, stream>>>(fkc, Wc, nWc, 1024);
    convert_split_kernel<<<65536, 256, 0, stream>>>(f1f, Af, nAf, 256);
    convert_split_kernel<<<65536, 256, 0, stream>>>(fkf, Wf, nWf, 256);
    dim3 gridC(32, 4);
    coarse_match_ws<<<gridC, 256, 0, stream>>>(Wc, Ac, nAc, outCoarse);
    fine_match_ws<<<4096, 128, 0, stream>>>(Af, Wf, nAf, outCoarse, outFine);
  } else {
    dim3 gridC(64, 4);
    coarse_match_direct<<<gridC, 256, 0, stream>>>(f1c, fkc, outCoarse);
    fine_match_direct<<<4096, 128, 0, stream>>>(f1f, fkf, outCoarse, outFine);
  }
}
